// SingleChannelVNTResnetWrapper_56212531970645
// MI455X (gfx1250) — matrix operand order repaired and checked by reading, not compiled, not run on hardware
//
#include <hip/hip_runtime.h>

#define BB   4
#define NN   4096
#define KK   20
#define COLS (3 * NN)  // 12288 columns per batch (n*3 + v)
#define PCH  32        // pool reduction chunks (NN / PCH = 128 points per chunk)

typedef _Float16 v16h __attribute__((ext_vector_type(16)));
typedef _Float16 v8h  __attribute__((ext_vector_type(8)));
typedef float    v8f  __attribute__((ext_vector_type(8)));

// ---------------------------------------------------------------------------
// Utility: f32 -> f16 weight conversion
// ---------------------------------------------------------------------------
__global__ void f32_to_f16(const float* __restrict__ s, _Float16* __restrict__ d, int n) {
  int i = blockIdx.x * blockDim.x + threadIdx.x;
  if (i < n) d[i] = (_Float16)s[i];
}

// ---------------------------------------------------------------------------
// kNN: per-point top-K (largest neg squared distance, includes self like jax)
// ---------------------------------------------------------------------------
__global__ void knn_topk(const float* __restrict__ pc, int* __restrict__ idx) {
  __shared__ float sp[256 * 3];
  const int b = blockIdx.y;
  const int n = blockIdx.x * 256 + threadIdx.x;
  const float px = pc[((size_t)b * NN + n) * 3 + 0];
  const float py = pc[((size_t)b * NN + n) * 3 + 1];
  const float pz = pc[((size_t)b * NN + n) * 3 + 2];

  float bd[KK];
  int   bi[KK];
#pragma unroll
  for (int s = 0; s < KK; ++s) { bd[s] = -3.4e38f; bi[s] = 0; }

  for (int mt = 0; mt < NN; mt += 256) {
    const int m = mt + threadIdx.x;
    sp[threadIdx.x * 3 + 0] = pc[((size_t)b * NN + m) * 3 + 0];
    sp[threadIdx.x * 3 + 1] = pc[((size_t)b * NN + m) * 3 + 1];
    sp[threadIdx.x * 3 + 2] = pc[((size_t)b * NN + m) * 3 + 2];
    __syncthreads();
    for (int j = 0; j < 256; ++j) {
      const float dx = sp[j * 3 + 0] - px;
      const float dy = sp[j * 3 + 1] - py;
      const float dz = sp[j * 3 + 2] - pz;
      const float val = -(dx * dx + dy * dy + dz * dz);
      if (val > bd[KK - 1]) {
        bd[KK - 1] = val; bi[KK - 1] = mt + j;
#pragma unroll
        for (int s = KK - 1; s > 0; --s) {  // one bubble pass restores order
          if (bd[s] > bd[s - 1]) {
            float tf = bd[s]; bd[s] = bd[s - 1]; bd[s - 1] = tf;
            int   ti = bi[s]; bi[s] = bi[s - 1]; bi[s - 1] = ti;
          }
        }
      }
    }
    __syncthreads();
  }
#pragma unroll
  for (int s = 0; s < KK; ++s) idx[((size_t)b * NN + n) * KK + s] = bi[s];
}

// ---------------------------------------------------------------------------
// Edge features + conv (Wf) + VN-leaky (Wd) + VN max-pool over k (pool_pos_Wd)
// One block per (b, n); 64 threads = 64 output channels. Writes column-major
// f16 activations y64[b][col=3n+v][64].
// ---------------------------------------------------------------------------
__global__ void edge_conv(const float* __restrict__ pc, const int* __restrict__ idx,
                          const float* __restrict__ Wf, const float* __restrict__ Wd,
                          const float* __restrict__ Wp, _Float16* __restrict__ y64) {
  __shared__ float snet[64 * 3 * KK];  // [c][v][k]
  const int b = blockIdx.y, n = blockIdx.x, t = threadIdx.x;
  const float p0 = pc[((size_t)b * NN + n) * 3 + 0];
  const float p1 = pc[((size_t)b * NN + n) * 3 + 1];
  const float p2 = pc[((size_t)b * NN + n) * 3 + 2];
  const float f0 = Wf[t * 3 + 0], f1 = Wf[t * 3 + 1], f2 = Wf[t * 3 + 2];
  const float g0 = Wd[t * 3 + 0], g1 = Wd[t * 3 + 1], g2 = Wd[t * 3 + 2];

  for (int k = 0; k < KK; ++k) {
    const int ii = idx[((size_t)b * NN + n) * KK + k];
    const float q0 = pc[((size_t)b * NN + ii) * 3 + 0];
    const float q1 = pc[((size_t)b * NN + ii) * 3 + 1];
    const float q2 = pc[((size_t)b * NN + ii) * 3 + 2];
    const float e0 = q0 - p0, e1 = q1 - p1, e2 = q2 - p2;
    const float c0 = q1 * p2 - q2 * p1;          // cross(nbr, x)
    const float c1 = q2 * p0 - q0 * p2;
    const float c2 = q0 * p1 - q1 * p0;
    float x0 = f0 * e0 + f1 * p0 + f2 * c0;
    float x1 = f0 * e1 + f1 * p1 + f2 * c1;
    float x2 = f0 * e2 + f1 * p2 + f2 * c2;
    const float d0 = g0 * e0 + g1 * p0 + g2 * c0;
    const float d1 = g0 * e1 + g1 * p1 + g2 * c1;
    const float d2 = g0 * e2 + g1 * p2 + g2 * c2;
    const float dot = x0 * d0 + x1 * d1 + x2 * d2;
    const float dsq = d0 * d0 + d1 * d1 + d2 * d2;
    if (dot < 0.0f) {
      const float fac = dot / (dsq + 1e-6f);
      x0 -= fac * d0; x1 -= fac * d1; x2 -= fac * d2;
    }
    snet[(t * 3 + 0) * KK + k] = x0;
    snet[(t * 3 + 1) * KK + k] = x1;
    snet[(t * 3 + 2) * KK + k] = x2;
  }
  __syncthreads();

  float best = -3.4e38f; int bk = 0;
  for (int k = 0; k < KK; ++k) {
    float d0 = 0.f, d1 = 0.f, d2 = 0.f;
    for (int c = 0; c < 64; ++c) {
      const float w = Wp[t * 64 + c];
      d0 += w * snet[(c * 3 + 0) * KK + k];
      d1 += w * snet[(c * 3 + 1) * KK + k];
      d2 += w * snet[(c * 3 + 2) * KK + k];
    }
    const float dot = snet[(t * 3 + 0) * KK + k] * d0 +
                      snet[(t * 3 + 1) * KK + k] * d1 +
                      snet[(t * 3 + 2) * KK + k] * d2;
    if (dot > best) { best = dot; bk = k; }
  }
#pragma unroll
  for (int v = 0; v < 3; ++v)
    y64[((size_t)b * COLS + 3 * n + v) * 64 + t] = (_Float16)snet[(t * 3 + v) * KK + bk];
}

// ---------------------------------------------------------------------------
// WMMA GEMM: Out[b][col][Cout] = W[Cout,CIN] @ X[b][col][CIN]   (f16 in, f32 acc)
// Register-blocked: one wave owns one 16-column tile and FOUR 16-row tiles
// (64 output channels). The activation (B) fragment is loaded once per K-step
// and reused across 4 WMMAs -> 4x less activation traffic, 4 wmma per K-step.
// A fragment (lane L): row = L&15, K-halves by L>>4 (contiguous k = 16*(L>>4) + e).
// B fragment (lane L): col = L&15, K-half  by L>>4 (same contiguous k order as A).
// D fragment (lane L): rows (L>>4)*8 + r, col = L&15.
// ---------------------------------------------------------------------------
#define ROWT 4  // 16-row tiles per wave; grid.y = Cout / (16*ROWT)

template <int CIN>
__global__ __launch_bounds__(256) void vn_gemm(const _Float16* __restrict__ W,
                                               const _Float16* __restrict__ X,
                                               const _Float16* __restrict__ addsrc,
                                               _Float16* __restrict__ Out, int Cout) {
  const int lane = threadIdx.x & 31;
  const int wave = threadIdx.x >> 5;
  const int b = blockIdx.z;
  const int colbase = blockIdx.x * 128 + wave * 16;
  const int ot = blockIdx.y * (16 * ROWT);
  const int nidx = lane & 15;
  const int hsel = lane >> 4;

  const _Float16* wp = W + ((size_t)(ot + nidx)) * CIN + hsel * 16;
  const _Float16* xp = X + ((size_t)b * COLS + colbase + nidx) * CIN + hsel * 16;

  v8f acc[ROWT] = {};
#pragma unroll
  for (int k0 = 0; k0 < CIN; k0 += 32) {
    const v16h bt = *(const v16h*)(xp + k0);  // activation fragment, loaded once
    if (k0 + 32 < CIN) __builtin_prefetch((const void*)(xp + k0 + 32), 0, 0);
#pragma unroll
    for (int rt = 0; rt < ROWT; ++rt) {
      const _Float16* wpr = wp + (size_t)rt * 16 * CIN + k0;
      const v8h alo = *(const v8h*)(wpr);        // K = kb .. kb+7
      const v8h ahi = *(const v8h*)(wpr + 8);    // K = kb+8 .. kb+15
      v16h a;
#pragma unroll
      for (int e = 0; e < 8; ++e) { a[e] = alo[e]; a[8 + e] = ahi[e]; }
      acc[rt] = __builtin_amdgcn_wmma_f32_16x16x32_f16(false, a, false, bt,
                                                       (short)0, acc[rt], false, false);
    }
  }

  const size_t colb = ((size_t)b * COLS + colbase + nidx) * (size_t)Cout;
#pragma unroll
  for (int rt = 0; rt < ROWT; ++rt) {
    const size_t cb = colb + ot + rt * 16 + hsel * 8;
    if (addsrc) {
      const _Float16* ap = addsrc + cb;
#pragma unroll
      for (int r = 0; r < 8; ++r) Out[cb + r] = (_Float16)(acc[rt][r] + (float)ap[r]);
    } else {
#pragma unroll
      for (int r = 0; r < 8; ++r) Out[cb + r] = (_Float16)acc[rt][r];
    }
  }
}

// ---------------------------------------------------------------------------
// VN leaky ReLU (ns=0): couples the 3 vector components of channel c at point n
// ---------------------------------------------------------------------------
__global__ void vn_leaky_k(const _Float16* __restrict__ x, const _Float16* __restrict__ d,
                           _Float16* __restrict__ out, int C) {
  const size_t t = (size_t)blockIdx.x * blockDim.x + threadIdx.x;
  const size_t total = (size_t)BB * C * NN;
  if (t >= total) return;
  const int c = (int)(t % C);
  const size_t r = t / C;
  const int n = (int)(r % NN);
  const int b = (int)(r / NN);
  const size_t base = ((size_t)b * COLS + 3 * n) * (size_t)C + c;
  float x0 = x[base], x1 = x[base + C], x2 = x[base + 2 * C];
  const float d0 = d[base], d1 = d[base + C], d2 = d[base + 2 * C];
  const float dot = x0 * d0 + x1 * d1 + x2 * d2;
  const float dsq = d0 * d0 + d1 * d1 + d2 * d2;
  if (dot < 0.0f) {
    const float fac = dot / (dsq + 1e-6f);
    x0 -= fac * d0; x1 -= fac * d1; x2 -= fac * d2;
  }
  out[base] = (_Float16)x0;
  out[base + C] = (_Float16)x1;
  out[base + 2 * C] = (_Float16)x2;
}

// ---------------------------------------------------------------------------
// VN max-pool over N, two-stage argmax of <net, dir>.
// Stage 1: grid (PCH, B) x 128 threads; each thread scans NN/PCH points and
// writes its chunk winner (dot, vec3) -> part[b][chunk][c][4].
// Stage 2: grid B x 128 threads; reduce chunks in order (first-index tie-break).
// ---------------------------------------------------------------------------
__global__ void vn_pool_part(const _Float16* __restrict__ net,
                             const _Float16* __restrict__ dir,
                             float* __restrict__ part) {
  const int chunk = blockIdx.x;
  const int b = blockIdx.y;
  const int c = threadIdx.x;  // 128
  const int n0 = chunk * (NN / PCH);
  float best = -3.4e38f, bv0 = 0.f, bv1 = 0.f, bv2 = 0.f;
  for (int n = n0; n < n0 + (NN / PCH); ++n) {
    const size_t base = ((size_t)b * COLS + 3 * n) * 128 + c;
    const float x0 = net[base], x1 = net[base + 128], x2 = net[base + 256];
    const float d0 = dir[base], d1 = dir[base + 128], d2 = dir[base + 256];
    const float dot = x0 * d0 + x1 * d1 + x2 * d2;
    if (dot > best) { best = dot; bv0 = x0; bv1 = x1; bv2 = x2; }
  }
  float* p = part + (((size_t)b * PCH + chunk) * 128 + c) * 4;
  p[0] = best; p[1] = bv0; p[2] = bv1; p[3] = bv2;
}

__global__ void vn_pool_final(const float* __restrict__ part, float* __restrict__ pooled) {
  const int b = blockIdx.x;
  const int c = threadIdx.x;  // 128
  float best = -3.4e38f, bv0 = 0.f, bv1 = 0.f, bv2 = 0.f;
  for (int chunk = 0; chunk < PCH; ++chunk) {  // in order -> first-index tie-break
    const float* p = part + (((size_t)b * PCH + chunk) * 128 + c) * 4;
    const float dot = p[0];
    if (dot > best) { best = dot; bv0 = p[1]; bv1 = p[2]; bv2 = p[3]; }
  }
  pooled[((size_t)b * 128 + c) * 3 + 0] = bv0;
  pooled[((size_t)b * 128 + c) * 3 + 1] = bv1;
  pooled[((size_t)b * 128 + c) * 3 + 2] = bv2;
}

// ---------------------------------------------------------------------------
// Concat [net(128) | broadcast(pooled)(128)] -> 256-channel column-major input
// ---------------------------------------------------------------------------
__global__ void vn_concat(const _Float16* __restrict__ net, const float* __restrict__ pooled,
                          _Float16* __restrict__ xout) {
  const size_t t = (size_t)blockIdx.x * 256 + threadIdx.x;  // B*N*256
  const int c = (int)(t & 255);
  const size_t r = t >> 8;
  const int n = (int)(r % NN);
  const int b = (int)(r / NN);
#pragma unroll
  for (int v = 0; v < 3; ++v) {
    _Float16 val;
    if (c < 128) val = net[((size_t)b * COLS + 3 * n + v) * 128 + c];
    else         val = (_Float16)pooled[((size_t)b * 128 + (c - 128)) * 3 + v];
    xout[((size_t)b * COLS + 3 * n + v) * 256 + c] = val;
  }
}

// ---------------------------------------------------------------------------
// Head: actc direction + leaky, then fcc projection. Tiny (12 columns total).
// ---------------------------------------------------------------------------
__global__ void head_act(const float* __restrict__ actc, const float* __restrict__ netf,
                         float* __restrict__ yh) {
  const int t = threadIdx.x;  // 512 = B*128
  const int b = t >> 7, c = t & 127;
  float d0 = 0.f, d1 = 0.f, d2 = 0.f;
  for (int c2 = 0; c2 < 128; ++c2) {
    const float w = actc[c * 128 + c2];
    d0 += w * netf[((size_t)b * 128 + c2) * 3 + 0];
    d1 += w * netf[((size_t)b * 128 + c2) * 3 + 1];
    d2 += w * netf[((size_t)b * 128 + c2) * 3 + 2];
  }
  float x0 = netf[((size_t)b * 128 + c) * 3 + 0];
  float x1 = netf[((size_t)b * 128 + c) * 3 + 1];
  float x2 = netf[((size_t)b * 128 + c) * 3 + 2];
  const float dot = x0 * d0 + x1 * d1 + x2 * d2;
  const float dsq = d0 * d0 + d1 * d1 + d2 * d2;
  if (dot < 0.0f) {
    const float fac = dot / (dsq + 1e-6f);
    x0 -= fac * d0; x1 -= fac * d1; x2 -= fac * d2;
  }
  yh[((size_t)b * 128 + c) * 3 + 0] = x0;
  yh[((size_t)b * 128 + c) * 3 + 1] = x1;
  yh[((size_t)b * 128 + c) * 3 + 2] = x2;
}

__global__ void head_fc(const float* __restrict__ fcc, const float* __restrict__ yh,
                        float* __restrict__ out) {
  const int t = threadIdx.x;  // 256 = B*64
  const int b = t >> 6, o = t & 63;
  float a0 = 0.f, a1 = 0.f, a2 = 0.f;
  for (int c = 0; c < 128; ++c) {
    const float w = fcc[o * 128 + c];
    a0 += w * yh[((size_t)b * 128 + c) * 3 + 0];
    a1 += w * yh[((size_t)b * 128 + c) * 3 + 1];
    a2 += w * yh[((size_t)b * 128 + c) * 3 + 2];
  }
  out[b * 192 + o * 3 + 0] = a0;
  out[b * 192 + o * 3 + 1] = a1;
  out[b * 192 + o * 3 + 2] = a2;
}

// ---------------------------------------------------------------------------
// Launch
// ---------------------------------------------------------------------------
extern "C" void kernel_launch(void* const* d_in, const int* in_sizes, int n_in,
                              void* d_out, int out_size, void* d_ws, size_t ws_size,
                              hipStream_t stream) {
  const float* pc       = (const float*)d_in[0];
  const float* conv_Wf  = (const float*)d_in[1];
  const float* conv_Wd  = (const float*)d_in[2];
  const float* pool_pos = (const float*)d_in[3];
  const float* fc_pos   = (const float*)d_in[4];
  const float* a0       = (const float*)d_in[5];
  const float* f0       = (const float*)d_in[6];
  const float* a1       = (const float*)d_in[7];
  const float* f1       = (const float*)d_in[8];
  const float* sc       = (const float*)d_in[9];
  const float* poolW    = (const float*)d_in[10];
  const float* actc     = (const float*)d_in[11];
  const float* fcc      = (const float*)d_in[12];
  (void)in_sizes; (void)n_in; (void)out_size; (void)ws_size;

  char* ws = (char*)d_ws;
  size_t off = 0;
  auto alloc = [&](size_t bytes) -> char* {
    off = (off + 255) & ~(size_t)255;
    char* p = ws + off;
    off += bytes;
    return p;
  };

  int*      p_idx   = (int*)alloc((size_t)BB * NN * KK * sizeof(int));
  _Float16* w_fcpos = (_Float16*)alloc((size_t)256 * 64 * 2);
  _Float16* w_a0    = (_Float16*)alloc((size_t)5 * 256 * 256 * 2);
  _Float16* w_f0    = (_Float16*)alloc((size_t)5 * 128 * 256 * 2);
  _Float16* w_a1    = (_Float16*)alloc((size_t)5 * 128 * 128 * 2);
  _Float16* w_f1    = (_Float16*)alloc((size_t)5 * 128 * 128 * 2);
  _Float16* w_sc    = (_Float16*)alloc((size_t)5 * 128 * 256 * 2);
  _Float16* w_pl    = (_Float16*)alloc((size_t)5 * 128 * 128 * 2);
  _Float16* p_y64   = (_Float16*)alloc((size_t)BB * COLS * 64 * 2);
  _Float16* slab0   = (_Float16*)alloc((size_t)BB * COLS * 256 * 2);
  _Float16* slab1   = (_Float16*)alloc((size_t)BB * COLS * 256 * 2);
  _Float16* slab2   = (_Float16*)alloc((size_t)BB * COLS * 256 * 2);
  float*    p_part  = (float*)alloc((size_t)BB * PCH * 128 * 4 * sizeof(float));
  float*    p_pool  = (float*)alloc((size_t)BB * 128 * 3 * sizeof(float));
  float*    p_yh    = (float*)alloc((size_t)BB * 128 * 3 * sizeof(float));

  const size_t h128 = (size_t)BB * COLS * 128;
  _Float16* p_x  = slab0;         // block input, 256 ch
  _Float16* p_t  = slab1;         // a0@x, 256 ch
  _Float16* p_y  = slab2;         // leaky(x,t), 256 ch
  _Float16* p_n  = slab1;         // f0@y, 128 ch (slab1 lo; t dead)
  _Float16* p_t2 = slab1 + h128;  // a1@n, 128 ch (slab1 hi)
  _Float16* p_z  = slab2;         // leaky(n,t2), 128 ch (slab2 lo; y dead)
  _Float16* p_s  = slab2 + h128;  // sc@x, 128 ch (slab2 hi)
  _Float16* p_n2 = slab1;         // f1@z + s, 128 ch (slab1 lo; n,t2 dead)
  _Float16* p_d  = slab1 + h128;  // pool@n2, 128 ch (slab1 hi)

  auto conv16 = [&](const float* s, _Float16* d, int n) {
    f32_to_f16<<<(n + 255) / 256, 256, 0, stream>>>(s, d, n);
  };
  conv16(fc_pos, w_fcpos, 256 * 64);
  conv16(a0, w_a0, 5 * 256 * 256);
  conv16(f0, w_f0, 5 * 128 * 256);
  conv16(a1, w_a1, 5 * 128 * 128);
  conv16(f1, w_f1, 5 * 128 * 128);
  conv16(sc, w_sc, 5 * 128 * 256);
  conv16(poolW, w_pl, 5 * 128 * 128);

  knn_topk<<<dim3(NN / 256, BB), 256, 0, stream>>>(pc, p_idx);
  edge_conv<<<dim3(NN, BB), 64, 0, stream>>>(pc, p_idx, conv_Wf, conv_Wd, pool_pos, p_y64);

  // fc_pos: [256,64] @ [64, 3N]  -> x (256 ch)
  vn_gemm<64><<<dim3(COLS / 128, 256 / 64, BB), 256, 0, stream>>>(w_fcpos, p_y64, nullptr, p_x, 256);

  for (int i = 0; i < 5; ++i) {
    vn_gemm<256><<<dim3(COLS / 128, 4, BB), 256, 0, stream>>>(w_a0 + (size_t)i * 256 * 256, p_x, nullptr, p_t, 256);
    vn_leaky_k<<<16384, 256, 0, stream>>>(p_x, p_t, p_y, 256);
    vn_gemm<256><<<dim3(COLS / 128, 2, BB), 256, 0, stream>>>(w_f0 + (size_t)i * 128 * 256, p_y, nullptr, p_n, 128);
    vn_gemm<128><<<dim3(COLS / 128, 2, BB), 256, 0, stream>>>(w_a1 + (size_t)i * 128 * 128, p_n, nullptr, p_t2, 128);
    vn_leaky_k<<<8192, 256, 0, stream>>>(p_n, p_t2, p_z, 128);
    vn_gemm<256><<<dim3(COLS / 128, 2, BB), 256, 0, stream>>>(w_sc + (size_t)i * 128 * 256, p_x, nullptr, p_s, 128);
    vn_gemm<128><<<dim3(COLS / 128, 2, BB), 256, 0, stream>>>(w_f1 + (size_t)i * 128 * 128, p_z, p_s, p_n2, 128);
    vn_gemm<128><<<dim3(COLS / 128, 2, BB), 256, 0, stream>>>(w_pl + (size_t)i * 128 * 128, p_n2, nullptr, p_d, 128);
    vn_pool_part<<<dim3(PCH, BB), 128, 0, stream>>>(p_n2, p_d, p_part);
    vn_pool_final<<<BB, 128, 0, stream>>>(p_part, p_pool);
    if (i < 4) vn_concat<<<16384, 256, 0, stream>>>(p_n2, p_pool, p_x);
  }

  head_act<<<1, 512, 0, stream>>>(actc, p_pool, p_yh);
  head_fc<<<1, 256, 0, stream>>>(fcc, p_yh, (float*)d_out);
}
